// Transformer2d_90821378441558
// MI455X (gfx1250) — matrix operand order repaired and checked by reading, not compiled, not run on hardware
//
#include <hip/hip_runtime.h>
#include <hip/hip_bf16.h>

// ---------------------------------------------------------------------------
// MI455X (gfx1250) transformer block.
// - All GEMMs: v_wmma_f32_16x16x32_f16, 16x64 macro-tile per wave (A reuse x4),
//   software double-buffered fragments (next-K loads overlap current WMMAs).
// - Attention: [Q;PE]^T[K;PE] bias fold, flash softmax, P staged via LDS,
//   V chunk staged via Tensor Data Mover (tensor_load_to_lds + tensorcnt),
//   next K chunk prefetched via global_prefetch.
// ---------------------------------------------------------------------------

typedef __attribute__((ext_vector_type(16))) _Float16 v16h;
typedef __attribute__((ext_vector_type(8)))  _Float16 v8h;
typedef __attribute__((ext_vector_type(8)))  float    v8f;
typedef __attribute__((ext_vector_type(4)))  unsigned u32x4;
typedef __attribute__((ext_vector_type(8)))  int      i32x8;
typedef __attribute__((ext_vector_type(4)))  int      i32x4;

#define Bsz   4
#define Cch   256
#define Hh    8
#define DH    32
#define DHP   64          // dh padded with posenc channels
#define HW    48
#define Ntok  2304        // 48*48
#define NB16  144         // Ntok/16
#define NB64  36          // Ntok/64

#if __has_builtin(__builtin_amdgcn_tensor_load_to_lds)
#  define HAS_TDM 1
#  if __has_include(<hip/amd_detail/amd_gfx1250_TDM.h>)
#    define TDM_ARGS6 1   // therock clang-23 headers -> 6-arg builtin
#  else
#    define TDM_ARGS6 0   // ROCm 7.2 clang-22 -> 5-arg builtin
#  endif
#else
#  define HAS_TDM 0
#endif

__device__ __forceinline__ v8f wmma16(v16h a, v16h b, v8f c) {
  return __builtin_amdgcn_wmma_f32_16x16x32_f16(false, a, false, b, (short)0, c,
                                                false, false);
}

// A fragment (16x32 f16, MxK). lane<16 -> row=lane, K 0..15;
// lane>=16 -> row=lane-16, K 16..31.
__device__ __forceinline__ v16h afrag(const _Float16* W, int ld, int lane) {
  int m  = lane & 15;
  int kb = (lane >> 4) * 16;
  const _Float16* p = W + m * ld + kb;
  return *(const v16h*)(p);
}

// B fragment (32x16 f16, KxN). lane<16 -> col=lane, K 0..15; lane>=16 ->
// col=lane-16, K 16..31. Column-major storage (K contiguous).
__device__ __forceinline__ v16h bfrag(const _Float16* Bc, int ld, int lane) {
  int n  = lane & 15;
  int kb = (lane >> 4) * 16;
  return *(const v16h*)(Bc + n * ld + kb);
}

// 16x64 output tile GEMM core, double-buffered fragment loads:
// fragments for step k+32 are fetched into fresh registers before the four
// WMMAs of step k issue, so VMEM latency overlaps the matrix pipe.
__device__ __forceinline__ void gemm16x64(const _Float16* A, int lda,
                                          const _Float16* Bb, int ldb,
                                          int Kdim, int lane, v8f acc[4]) {
  v16h ac  = afrag(A, lda, lane);
  v16h bc0 = bfrag(Bb + 0 * 16 * ldb, ldb, lane);
  v16h bc1 = bfrag(Bb + 1 * 16 * ldb, ldb, lane);
  v16h bc2 = bfrag(Bb + 2 * 16 * ldb, ldb, lane);
  v16h bc3 = bfrag(Bb + 3 * 16 * ldb, ldb, lane);
#pragma unroll
  for (int k = 0; k < Kdim; k += 32) {
    v16h an = ac, bn0 = bc0, bn1 = bc1, bn2 = bc2, bn3 = bc3;
    if (k + 32 < Kdim) {
      an  = afrag(A + k + 32, lda, lane);
      bn0 = bfrag(Bb + 0 * 16 * ldb + k + 32, ldb, lane);
      bn1 = bfrag(Bb + 1 * 16 * ldb + k + 32, ldb, lane);
      bn2 = bfrag(Bb + 2 * 16 * ldb + k + 32, ldb, lane);
      bn3 = bfrag(Bb + 3 * 16 * ldb + k + 32, ldb, lane);
    }
    acc[0] = wmma16(ac, bc0, acc[0]);
    acc[1] = wmma16(ac, bc1, acc[1]);
    acc[2] = wmma16(ac, bc2, acc[2]);
    acc[3] = wmma16(ac, bc3, acc[3]);
    ac = an; bc0 = bn0; bc1 = bn1; bc2 = bn2; bc3 = bn3;
  }
}

// ---------------------------------------------------------------------------
// 1. fp32 weights -> f16
__global__ void k_convert_w(const float* wq, const float* w1, const float* w2,
                            _Float16* wq16, _Float16* w116, _Float16* w216) {
  int i = blockIdx.x * 256 + threadIdx.x;                 // 720896 total, exact
  if (i < 196608)             wq16[i]         = (_Float16)wq[i];
  else if (i < 196608+262144) w116[i-196608]  = (_Float16)w1[i-196608];
  else                        w216[i-458752]  = (_Float16)w2[i-458752];
}

// 2. posenc -> padded channels [32,64) of q_pad / k_pad
__global__ void k_posenc(_Float16* qpad, _Float16* kpad) {
  int i  = blockIdx.x * 256 + threadIdx.x;   // B*Hh*Ntok*DH = 2359296, exact
  int kk = i & 31;
  int n  = (i >> 5) % Ntok;
  int bh = (i >> 5) / Ntok;                  // 0..31
  int h  = bh & 7;
  int cg  = h * DH + kk;
  int g   = cg >> 2;
  int rem = cg & 3;
  int hh = n / HW, ww = n % HW;
  float f   = __powf(10000.0f, -(float)g * (1.0f / 64.0f));
  float pos = (rem < 2) ? (float)hh : (float)ww;
  float ph  = f * pos;
  float val = (rem & 1) ? __sinf(ph) : __cosf(ph);
  int addr = (bh * Ntok + n) * DHP + 32 + kk;
  qpad[addr] = (_Float16)val;
  kpad[addr] = (_Float16)val;
}

// 3. LayerNorm over C, one wave per token -> token-major f16 [b][n][c]
__global__ void k_ln1(const float* x, const float* w, const float* bb,
                      _Float16* y) {
  int lane = threadIdx.x & 31, wv = threadIdx.x >> 5;
  int t = blockIdx.x * 8 + wv;               // B*Ntok = 9216 tokens, exact
  int b = t / Ntok, n = t % Ntok;
  float vals[8]; float s = 0.f, s2 = 0.f;
#pragma unroll
  for (int i = 0; i < 8; ++i) {
    int c = lane + 32 * i;
    float v = x[(b * Cch + c) * Ntok + n];
    vals[i] = v; s += v; s2 += v * v;
  }
#pragma unroll
  for (int m = 1; m < 32; m <<= 1) { s += __shfl_xor(s, m); s2 += __shfl_xor(s2, m); }
  float mu = s * (1.0f / Cch);
  float rs = rsqrtf(s2 * (1.0f / Cch) - mu * mu + 1e-5f);
#pragma unroll
  for (int i = 0; i < 8; ++i) {
    int c = lane + 32 * i;
    y[(b * Ntok + n) * Cch + c] = (_Float16)((vals[i] - mu) * rs * w[c] + bb[c]);
  }
}

// 4. QKV GEMM, 16x64 tile/wave: out[o][n] = sum_c W[o][c]*Y[n][c], route q/k/v
__global__ void k_gemm_qkv(const _Float16* W, const _Float16* Y, const float* bias,
                           _Float16* qpad, _Float16* kpad, _Float16* vbuf) {
  int lane = threadIdx.x & 31, wv = threadIdx.x >> 5;
  int tile = blockIdx.x * 8 + wv;            // 4*48*36 = 6912 tiles, exact
  int b = tile / (48 * NB64); int rem = tile % (48 * NB64);
  int ob = rem / NB64, nb = rem % NB64;
  v8f acc[4] = {};
  gemm16x64(W + ob * 16 * Cch, Cch,
            Y + (b * Ntok + nb * 64) * Cch, Cch, Cch, lane, acc);
  int half = lane >> 4, col = lane & 15;
#pragma unroll
  for (int j = 0; j < 4; ++j) {
    int n = nb * 64 + j * 16 + col;
#pragma unroll
    for (int r = 0; r < 8; ++r) {
      int o = ob * 16 + r + half * 8;
      float val = acc[j][r] + bias[o];
      int sect = o >> 8, oo = o & 255, h = oo >> 5, cc = oo & 31;
      if (sect == 0)      qpad[((b*Hh + h) * Ntok + n) * DHP + cc] = (_Float16)val;
      else if (sect == 1) kpad[((b*Hh + h) * Ntok + n) * DHP + cc] = (_Float16)val;
      else                vbuf[((b*Hh + h) * DH + cc) * Ntok + n]  = (_Float16)val;
    }
  }
}

// 5. Flash attention. One wave per 16-row block; 32 key-cols per step.
//    Scores: 4 WMMA (K=64). V chunk staged via TDM into LDS. P via LDS. 6 WMMA.
__global__ void k_attn(const _Float16* qpad, const _Float16* kpad,
                       const _Float16* vbuf, float* attn) {
  __shared__ __align__(32) _Float16 plds[4][16 * 32];
#if HAS_TDM
  __shared__ __align__(16) _Float16 vlds[4][32 * 32];
#endif
  int lane = threadIdx.x & 31, wv = threadIdx.x >> 5;
  int tile = blockIdx.x * 4 + wv;            // 4*8*144 = 4608 tiles, exact
  int b = tile / (Hh * NB16); int rem = tile % (Hh * NB16);
  int h = rem / NB16, nb = rem % NB16;
  int bh = b * Hh + h;
  const _Float16* Q = qpad + ((long)bh * Ntok + nb * 16) * DHP;
  const _Float16* K = kpad + (long)bh * Ntok * DHP;
  const _Float16* V = vbuf + (long)bh * DH * Ntok;
  _Float16* pl = plds[wv];

#if HAS_TDM
  _Float16* vl = vlds[wv];
  // TDM descriptor group 1 (constant): workgroup_mask=0, data_size=2B,
  // tensor_dim0=2304, tensor_dim1=32, tile_dim0=32, tile_dim1=32,
  // tensor_dim0_stride=2304 (packed per D# group-1 bit layout).
  unsigned lds_base = (unsigned)(unsigned long long)
      (__attribute__((address_space(3))) void*)vl;
  i32x8 g1 = { 0x00010000, 0x09000000, 0x00200000, 0x00200000, 32, 2304, 0, 0 };
  i32x4 gz = { 0, 0, 0, 0 };
#if TDM_ARGS6
  i32x8 gz8 = { 0, 0, 0, 0, 0, 0, 0, 0 };
#endif
#endif

  v16h a0 = afrag(Q, DHP, lane);             // k-chunk 0..31
  v16h a1 = afrag(Q + 32, DHP, lane);        // k-chunk 32..63
  v8f o0 = {}, o1 = {};
  float M[8], L[8];
#pragma unroll
  for (int r = 0; r < 8; ++r) { M[r] = -__builtin_inff(); L[r] = 0.f; }
  const float scale = 1.0f / 16.0f;          // 1/sqrt(C)
  int half = lane >> 4, col = lane & 15;

  for (int mb = 0; mb < Ntok; mb += 32) {
#if HAS_TDM
    {  // launch TDM for this step's V chunk (32 rows x 32 keys, 2 KB) -> LDS
      unsigned long long ga = (unsigned long long)(const void*)(V + mb);
      u32x4 g0 = { 1u,                       // count=1 valid descriptor
                   lds_base,                 // LDS byte address
                   (unsigned)ga,             // global addr [31:0]
                   ((unsigned)(ga >> 32) & 0x01ffffffu) | 0x80000000u }; // type=2
#if TDM_ARGS6
      __builtin_amdgcn_tensor_load_to_lds(g0, g1, gz, gz, gz8, 0);
#else
      __builtin_amdgcn_tensor_load_to_lds(g0, g1, gz, gz, 0);
#endif
    }
#endif
    // prefetch next K chunk (4 KB contiguous; 32 lanes x 128 B)
    __builtin_prefetch(K + (long)(mb + 32) * DHP + lane * 64, 0, 0);

    const _Float16* K0 = K + (long)mb * DHP;
    const _Float16* K1 = K0 + 16 * DHP;
    v8f s0 = {}, s1 = {};
    s0 = wmma16(a0, bfrag(K0,      DHP, lane), s0);
    s0 = wmma16(a1, bfrag(K0 + 32, DHP, lane), s0);
    s1 = wmma16(a0, bfrag(K1,      DHP, lane), s1);
    s1 = wmma16(a1, bfrag(K1 + 32, DHP, lane), s1);

    float p0[8], p1[8];
#pragma unroll
    for (int r = 0; r < 8; ++r) {
      float e0 = s0[r] * scale, e1 = s1[r] * scale;
      float cm = fmaxf(e0, e1);
      cm = fmaxf(cm, __shfl_xor(cm, 1));
      cm = fmaxf(cm, __shfl_xor(cm, 2));
      cm = fmaxf(cm, __shfl_xor(cm, 4));
      cm = fmaxf(cm, __shfl_xor(cm, 8));     // row max within 16-lane group
      float nM = fmaxf(M[r], cm);
      float al = __expf(M[r] - nM);
      M[r] = nM;
      float q0 = __expf(e0 - nM), q1 = __expf(e1 - nM);
      p0[r] = q0; p1[r] = q1;
      float cs = q0 + q1;
      cs += __shfl_xor(cs, 1); cs += __shfl_xor(cs, 2);
      cs += __shfl_xor(cs, 4); cs += __shfl_xor(cs, 8);
      L[r] = L[r] * al + cs;
      o0[r] *= al; o1[r] *= al;
    }
    // stage P (16x32) to LDS so it can be re-read as an A fragment
#pragma unroll
    for (int r = 0; r < 8; ++r) {
      int row = r + half * 8;
      pl[row * 32 + col]      = (_Float16)p0[r];
      pl[row * 32 + col + 16] = (_Float16)p1[r];
    }
    asm volatile("s_wait_dscnt 0" ::: "memory");
    v16h pa = afrag(pl, 32, lane);
#if HAS_TDM
    __builtin_amdgcn_s_wait_tensorcnt((short)0);   // V chunk landed in LDS
    o0 = wmma16(pa, bfrag(vl,           32, lane), o0);
    o1 = wmma16(pa, bfrag(vl + 16 * 32, 32, lane), o1);
#else
    o0 = wmma16(pa, bfrag(V + mb,             Ntok, lane), o0);
    o1 = wmma16(pa, bfrag(V + 16 * Ntok + mb, Ntok, lane), o1);
#endif
    asm volatile("s_wait_dscnt 0" ::: "memory");   // LDS reads done before reuse
  }
#pragma unroll
  for (int r = 0; r < 8; ++r) {
    int n = nb * 16 + r + half * 8;
    float inv = 1.0f / L[r];
    attn[((long)b * Cch + h * DH + col)      * Ntok + n] = o0[r] * inv;
    attn[((long)b * Cch + h * DH + col + 16) * Ntok + n] = o1[r] * inv;
  }
}

// 6. residual (x1 = x + gamma1*attn) + LayerNorm -> token-major f16
__global__ void k_resid_ln2(const float* x, const float* attn, const float* g1,
                            const float* w, const float* bb,
                            float* x1, _Float16* y) {
  int lane = threadIdx.x & 31, wv = threadIdx.x >> 5;
  int t = blockIdx.x * 8 + wv;
  int b = t / Ntok, n = t % Ntok;
  float vals[8]; float s = 0.f, s2 = 0.f;
#pragma unroll
  for (int i = 0; i < 8; ++i) {
    int c = lane + 32 * i;
    int idx = (b * Cch + c) * Ntok + n;
    float v = x[idx] + g1[c] * attn[idx];
    x1[idx] = v;
    vals[i] = v; s += v; s2 += v * v;
  }
#pragma unroll
  for (int m = 1; m < 32; m <<= 1) { s += __shfl_xor(s, m); s2 += __shfl_xor(s2, m); }
  float mu = s * (1.0f / Cch);
  float rs = rsqrtf(s2 * (1.0f / Cch) - mu * mu + 1e-5f);
#pragma unroll
  for (int i = 0; i < 8; ++i) {
    int c = lane + 32 * i;
    y[(b * Ntok + n) * Cch + c] = (_Float16)((vals[i] - mu) * rs * w[c] + bb[c]);
  }
}

// 7. MLP expand GEMM (16x64 tile/wave) + exact GELU -> token-major f16
__global__ void k_gemm_mlp1(const _Float16* W1, const _Float16* Z, const float* b1,
                            _Float16* Hbuf) {
  int lane = threadIdx.x & 31, wv = threadIdx.x >> 5;
  int tile = blockIdx.x * 8 + wv;            // 4*64*36 = 9216 tiles, exact
  int b = tile / (64 * NB64); int rem = tile % (64 * NB64);
  int ob = rem / NB64, nb = rem % NB64;
  v8f acc[4] = {};
  gemm16x64(W1 + ob * 16 * Cch, Cch,
            Z + (b * Ntok + nb * 64) * Cch, Cch, Cch, lane, acc);
  int half = lane >> 4, col = lane & 15;
#pragma unroll
  for (int j = 0; j < 4; ++j) {
    int n = nb * 64 + j * 16 + col;
#pragma unroll
    for (int r = 0; r < 8; ++r) {
      int o = ob * 16 + r + half * 8;
      float v = acc[j][r] + b1[o];
      float g = 0.5f * v * (1.0f + erff(v * 0.70710678118f));   // exact GELU
      Hbuf[(b * Ntok + n) * 1024 + o] = (_Float16)g;
    }
  }
}

// 8. MLP contract GEMM (16x64 tile/wave) + residual -> fp32 output [b][c][n]
__global__ void k_gemm_mlp2(const _Float16* W2, const _Float16* Hbuf, const float* b2,
                            const float* x1, const float* g2, float* out) {
  int lane = threadIdx.x & 31, wv = threadIdx.x >> 5;
  int tile = blockIdx.x * 8 + wv;            // 4*16*36 = 2304 tiles, exact
  int b = tile / (16 * NB64); int rem = tile % (16 * NB64);
  int ob = rem / NB64, nb = rem % NB64;
  v8f acc[4] = {};
  gemm16x64(W2 + ob * 16 * 1024, 1024,
            Hbuf + (b * Ntok + nb * 64) * 1024, 1024, 1024, lane, acc);
  int half = lane >> 4, col = lane & 15;
#pragma unroll
  for (int j = 0; j < 4; ++j) {
    int n = nb * 64 + j * 16 + col;
#pragma unroll
    for (int r = 0; r < 8; ++r) {
      int o = ob * 16 + r + half * 8;
      int idx = (b * Cch + o) * Ntok + n;
      out[idx] = x1[idx] + g2[o] * (acc[j][r] + b2[o]);
    }
  }
}

// ---------------------------------------------------------------------------
extern "C" void kernel_launch(void* const* d_in, const int* in_sizes, int n_in,
                              void* d_out, int out_size, void* d_ws, size_t ws_size,
                              hipStream_t stream) {
  const float* x      = (const float*)d_in[0];
  const float* ln1_w  = (const float*)d_in[1];
  const float* ln1_b  = (const float*)d_in[2];
  const float* wqkv   = (const float*)d_in[3];
  const float* bqkv   = (const float*)d_in[4];
  const float* gamma1 = (const float*)d_in[5];
  const float* ln2_w  = (const float*)d_in[6];
  const float* ln2_b  = (const float*)d_in[7];
  const float* w1     = (const float*)d_in[8];
  const float* b1     = (const float*)d_in[9];
  const float* w2     = (const float*)d_in[10];
  const float* b2     = (const float*)d_in[11];
  const float* gamma2 = (const float*)d_in[12];
  float* out = (float*)d_out;

  char* p = (char*)d_ws;
  auto take = [&](size_t bytes) {
    void* r = (void*)p;
    p += (bytes + 255) & ~(size_t)255;
    return r;
  };
  _Float16* yln1 = (_Float16*)take((size_t)Bsz * Ntok * Cch * 2);
  _Float16* qpad = (_Float16*)take((size_t)Bsz * Hh * Ntok * DHP * 2);
  _Float16* kpad = (_Float16*)take((size_t)Bsz * Hh * Ntok * DHP * 2);
  _Float16* vbuf = (_Float16*)take((size_t)Bsz * Hh * DH * Ntok * 2);
  float*    attn = (float*)   take((size_t)Bsz * Cch * Ntok * 4);
  float*    x1   = (float*)   take((size_t)Bsz * Cch * Ntok * 4);
  _Float16* yln2 = (_Float16*)take((size_t)Bsz * Ntok * Cch * 2);
  _Float16* hbuf = (_Float16*)take((size_t)Bsz * Ntok * 1024 * 2);
  _Float16* wq16 = (_Float16*)take((size_t)768 * 256 * 2);
  _Float16* w116 = (_Float16*)take((size_t)1024 * 256 * 2);
  _Float16* w216 = (_Float16*)take((size_t)256 * 1024 * 2);

  k_convert_w <<<2816, 256, 0, stream>>>(wqkv, w1, w2, wq16, w116, w216);
  k_posenc    <<<9216, 256, 0, stream>>>(qpad, kpad);
  k_ln1       <<<1152, 256, 0, stream>>>(x, ln1_w, ln1_b, yln1);
  k_gemm_qkv  <<< 864, 256, 0, stream>>>(wq16, yln1, bqkv, qpad, kpad, vbuf);
  k_attn      <<<1152, 128, 0, stream>>>(qpad, kpad, vbuf, attn);
  k_resid_ln2 <<<1152, 256, 0, stream>>>(x, attn, gamma1, ln2_w, ln2_b, x1, yln2);
  k_gemm_mlp1 <<<1152, 256, 0, stream>>>(w116, yln2, b1, hbuf);
  k_gemm_mlp2 <<< 288, 256, 0, stream>>>(w216, hbuf, b2, x1, gamma2, out);
}
